// ConvexPolytopeManifold_3212635537606
// MI455X (gfx1250) — matrix operand order repaired and checked by reading, not compiled, not run on hardware
//
#include <hip/hip_runtime.h>

// Problem constants from the reference
#define BB 4096   // batch
#define NN 512    // n
#define MM 1024   // m
#define TOLV 1e-5f
#define STEP 0.01f

typedef __bf16 bf16;
typedef __attribute__((ext_vector_type(16))) __bf16 v16bf;
typedef __attribute__((ext_vector_type(8)))  __bf16 v8bf;
typedef __attribute__((ext_vector_type(8)))  float  v8f;

// ---------------------------------------------------------------------------
// Fragment loaders for V_WMMA_F32_16X16X32_BF16 (wave32 layouts, ISA 7.12.2):
//   A-matrix 16x32 bf16: lane L (row = L&15, half = L>>4) holds
//     elements e=0..7  -> K = k0 + 8*half + e        (contiguous 16B)
//     elements e=8..15 -> K = k0 + 16 + 8*half + (e-8) (contiguous 16B)
//   B-matrix 32x16 bf16: lane L (col = L&15, half = L>>4) holds
//     same K mapping as A (two contiguous 16B chunks)
// Weight ("B") operand stored [N, K] row-major -> contiguous per-lane loads.
// ---------------------------------------------------------------------------
__device__ __forceinline__ v16bf load_a_frag(const bf16* __restrict__ base, int lda,
                                             int row, int k0, int half) {
  const bf16* p = base + (size_t)row * lda + k0 + 8 * half;
  union { v16bf v; v8bf h[2]; } u;
  u.h[0] = *(const v8bf*)p;
  u.h[1] = *(const v8bf*)(p + 16);
  return u.v;
}

__device__ __forceinline__ v16bf load_b_frag(const bf16* __restrict__ base, int ldb,
                                             int col, int k0, int half) {
  const bf16* p = base + (size_t)col * ldb + k0 + 8 * half;
  union { v16bf v; v8bf h[2]; } u;
  u.h[0] = *(const v8bf*)p;
  u.h[1] = *(const v8bf*)(p + 16);
  return u.v;
}

__device__ __forceinline__ v8f wmma_bf16(v16bf a, v16bf b, v8f c) {
  return __builtin_amdgcn_wmma_f32_16x16x32_bf16(false, a, false, b, (short)0, c,
                                                 false, false);
}

// ---------------------------------------------------------------------------
// Generic NT GEMM: C[M,N] = sum_k Aop[M,k] * Bop[N,k], bf16 in / f32 accum.
// Block = 256 threads = 8 waves arranged 2(M) x 4(N); wave tile 64x32
// (4 M-frags x 2 N-frags, 8 WMMA per 32-wide K step). Block tile 128x128.
// Epilogue selected at compile time.
// ---------------------------------------------------------------------------
enum { EPI_QBF = 0, EPI_C = 1, EPI_PGD = 2, EPI_Z = 3, EPI_FINAL = 4 };

template <int EPI>
__global__ __launch_bounds__(256)
void gemm_bf16_nt(const bf16* __restrict__ Aop, const bf16* __restrict__ Bop,
                  int K, int lda, int ldb, int N,
                  float* __restrict__ outf, bf16* __restrict__ outbf,
                  float* __restrict__ lamf, const float* __restrict__ cvec,
                  const float* __restrict__ act,
                  const float* __restrict__ e0, const float* __restrict__ e1,
                  const float* __restrict__ bias) {
  const int lane  = threadIdx.x & 31;
  const int wave  = threadIdx.x >> 5;
  const int half  = lane >> 4;
  const int lcol  = lane & 15;
  const int waveM = wave >> 2;   // 0..1
  const int waveN = wave & 3;    // 0..3
  const int row0  = blockIdx.y * 128 + waveM * 64;
  const int col0  = blockIdx.x * 128 + waveN * 32;

  v8f acc[4][2] = {};

  for (int k0 = 0; k0 < K; k0 += 32) {
    if (k0 + 32 < K) {  // prefetch next weight chunk (global_prefetch_b8)
      __builtin_prefetch(Bop + (size_t)(col0 + lcol) * ldb + k0 + 32, 0, 0);
    }
    v16bf bfr0 = load_b_frag(Bop, ldb, col0 + lcol,      k0, half);
    v16bf bfr1 = load_b_frag(Bop, ldb, col0 + 16 + lcol, k0, half);
#pragma unroll
    for (int mi = 0; mi < 4; ++mi) {
      v16bf afr = load_a_frag(Aop, lda, row0 + mi * 16 + lcol, k0, half);
      acc[mi][0] = wmma_bf16(afr, bfr0, acc[mi][0]);
      acc[mi][1] = wmma_bf16(afr, bfr1, acc[mi][1]);
    }
  }

  // C/D layout: lane -> col = lcol, VGPR r -> row = r + 8*half
#pragma unroll
  for (int mi = 0; mi < 4; ++mi) {
#pragma unroll
    for (int ni = 0; ni < 2; ++ni) {
      const int col = col0 + ni * 16 + lcol;
#pragma unroll
      for (int r = 0; r < 8; ++r) {
        const int row = row0 + mi * 16 + r + 8 * half;
        const size_t idx = (size_t)row * N + col;
        const float v = acc[mi][ni][r];
        if constexpr (EPI == EPI_QBF) {
          outbf[idx] = (bf16)v;                       // Q = A A^T (bf16)
        } else if constexpr (EPI == EPI_C) {
          outf[idx] = v - bias[col];                  // c = S A^T - b
        } else if constexpr (EPI == EPI_PGD) {
          float nl = lamf[idx] - STEP * (v - cvec[idx]);
          nl = nl > 0.f ? nl : 0.f;
          if (act) nl *= act[idx];
          lamf[idx]  = nl;                            // fp32 state (in place)
          outbf[idx] = (bf16)nl;                      // bf16 copy for next GEMM
        } else if constexpr (EPI == EPI_Z) {
          outbf[idx] = (bf16)(e0[idx] + e1[idx] - v); // z = (x+u) - lam A
        } else {                                      // EPI_FINAL
          outf[idx] = e1[idx] - v;                    // out = u - lam A
        }
      }
    }
  }
}

// ---------------------------------------------------------------------------
// Dual GEMM for the active-set mask: Ax = z A^T and Au = u A^T share the
// weight fragments. Wave tile 32x32 (2x2 frags, two accumulator sets).
// Block = 8 waves as 2(M) x 4(N) -> 64 x 128 tile.
//   active = (Ax >= b - TOL); masked = Au * active
// ---------------------------------------------------------------------------
__global__ __launch_bounds__(256)
void gemm_mask(const bf16* __restrict__ Zbf, const bf16* __restrict__ Ubf,
               const bf16* __restrict__ Abf, int K, int ldab, int N,
               const float* __restrict__ bvec,
               float* __restrict__ actv, float* __restrict__ masked) {
  const int lane  = threadIdx.x & 31;
  const int wave  = threadIdx.x >> 5;
  const int half  = lane >> 4;
  const int lcol  = lane & 15;
  const int waveM = wave >> 2;
  const int waveN = wave & 3;
  const int row0  = blockIdx.y * 64 + waveM * 32;
  const int col0  = blockIdx.x * 128 + waveN * 32;

  v8f az[2][2] = {};
  v8f au[2][2] = {};

  for (int k0 = 0; k0 < K; k0 += 32) {
    v16bf bfr0 = load_b_frag(Abf, ldab, col0 + lcol,      k0, half);
    v16bf bfr1 = load_b_frag(Abf, ldab, col0 + 16 + lcol, k0, half);
#pragma unroll
    for (int mi = 0; mi < 2; ++mi) {
      v16bf zf = load_a_frag(Zbf, ldab, row0 + mi * 16 + lcol, k0, half);
      v16bf uf = load_a_frag(Ubf, ldab, row0 + mi * 16 + lcol, k0, half);
      az[mi][0] = wmma_bf16(zf, bfr0, az[mi][0]);
      az[mi][1] = wmma_bf16(zf, bfr1, az[mi][1]);
      au[mi][0] = wmma_bf16(uf, bfr0, au[mi][0]);
      au[mi][1] = wmma_bf16(uf, bfr1, au[mi][1]);
    }
  }

#pragma unroll
  for (int mi = 0; mi < 2; ++mi) {
#pragma unroll
    for (int ni = 0; ni < 2; ++ni) {
      const int col = col0 + ni * 16 + lcol;
#pragma unroll
      for (int r = 0; r < 8; ++r) {
        const int row = row0 + mi * 16 + r + 8 * half;
        const size_t idx = (size_t)row * N + col;
        const float a = (az[mi][ni][r] >= bvec[col] - TOLV) ? 1.f : 0.f;
        actv[idx]   = a;
        masked[idx] = au[mi][ni][r] * a;
      }
    }
  }
}

// ---------------------------------------------------------------------------
// Elementwise prep kernels
// ---------------------------------------------------------------------------
__global__ void prep_A(const float* __restrict__ A, bf16* __restrict__ Abf,
                       bf16* __restrict__ ATbf) {
  int i = blockIdx.x * blockDim.x + threadIdx.x;
  if (i < MM * NN) {
    int r = i / NN, c = i % NN;
    float v = A[i];
    Abf[i] = (bf16)v;
    ATbf[(size_t)c * MM + r] = (bf16)v;  // A^T [n, m] for lam @ A
  }
}

__global__ void prep_S(const float* __restrict__ x, const float* __restrict__ u,
                       bf16* __restrict__ Sbf, bf16* __restrict__ Ubf) {
  int i = blockIdx.x * blockDim.x + threadIdx.x;
  if (i < BB * NN) {
    Sbf[i] = (bf16)(x[i] + u[i]);
    Ubf[i] = (bf16)u[i];
  }
}

__global__ void zero_lam(float* __restrict__ lamf, bf16* __restrict__ lambf) {
  int i = blockIdx.x * blockDim.x + threadIdx.x;
  if (i < BB * MM) {
    lamf[i]  = 0.f;
    lambf[i] = (bf16)0.f;
  }
}

// ---------------------------------------------------------------------------
extern "C" void kernel_launch(void* const* d_in, const int* in_sizes, int n_in,
                              void* d_out, int out_size, void* d_ws, size_t ws_size,
                              hipStream_t stream) {
  (void)in_sizes; (void)n_in; (void)out_size; (void)ws_size;
  const float* x = (const float*)d_in[0];
  const float* u = (const float*)d_in[1];
  const float* A = (const float*)d_in[2];
  const float* b = (const float*)d_in[3];
  float* out = (float*)d_out;

  // Carve workspace (~80 MB total)
  char* w = (char*)d_ws;
  size_t off = 0;
  auto carve = [&](size_t bytes) -> void* {
    void* p = w + off;
    off += (bytes + 255) & ~(size_t)255;
    return p;
  };
  bf16*  Abf   = (bf16*)carve((size_t)MM * NN * 2);
  bf16*  ATbf  = (bf16*)carve((size_t)MM * NN * 2);
  bf16*  Qbf   = (bf16*)carve((size_t)MM * MM * 2);
  bf16*  Sbf   = (bf16*)carve((size_t)BB * NN * 2);
  bf16*  Ubf   = (bf16*)carve((size_t)BB * NN * 2);
  bf16*  Zbf   = (bf16*)carve((size_t)BB * NN * 2);
  bf16*  lamb0 = (bf16*)carve((size_t)BB * MM * 2);
  bf16*  lamb1 = (bf16*)carve((size_t)BB * MM * 2);
  float* lamf  = (float*)carve((size_t)BB * MM * 4);
  float* cbuf  = (float*)carve((size_t)BB * MM * 4);  // c, then reused as "masked"
  float* actb  = (float*)carve((size_t)BB * MM * 4);

  prep_A<<<(MM * NN + 255) / 256, 256, 0, stream>>>(A, Abf, ATbf);
  prep_S<<<(BB * NN + 255) / 256, 256, 0, stream>>>(x, u, Sbf, Ubf);

  // Q = A A^T  (bf16 output for the PGD GEMMs)
  gemm_bf16_nt<EPI_QBF><<<dim3(MM / 128, MM / 128), 256, 0, stream>>>(
      Abf, Abf, NN, NN, NN, MM, nullptr, Qbf, nullptr, nullptr, nullptr,
      nullptr, nullptr, nullptr);

  // c = S A^T - b
  gemm_bf16_nt<EPI_C><<<dim3(MM / 128, BB / 128), 256, 0, stream>>>(
      Sbf, Abf, NN, NN, NN, MM, cbuf, nullptr, nullptr, nullptr, nullptr,
      nullptr, nullptr, b);

  zero_lam<<<(BB * MM + 255) / 256, 256, 0, stream>>>(lamf, lamb0);

  bf16* lp[2] = {lamb0, lamb1};
  int p = 0;
  // Phase 1: 50 iterations of lam <- relu(lam - step*(lam Q - c))
  for (int it = 0; it < 50; ++it) {
    gemm_bf16_nt<EPI_PGD><<<dim3(MM / 128, BB / 128), 256, 0, stream>>>(
        lp[p], Qbf, MM, MM, MM, MM, nullptr, lp[1 - p], lamf, cbuf, nullptr,
        nullptr, nullptr, nullptr);
    p ^= 1;
  }
  // p == 0 here (50 flips): latest lam in lp[0]

  // z = (x + u) - lam A   (bf16)
  gemm_bf16_nt<EPI_Z><<<dim3(NN / 128, BB / 128), 256, 0, stream>>>(
      lp[p], ATbf, MM, MM, MM, NN, nullptr, Zbf, nullptr, nullptr, nullptr,
      x, u, nullptr);

  // active mask + masked = (u A^T) * active   (masked overwrites cbuf)
  gemm_mask<<<dim3(MM / 128, BB / 64), 256, 0, stream>>>(
      Zbf, Ubf, Abf, NN, NN, MM, b, actb, cbuf);

  zero_lam<<<(BB * MM + 255) / 256, 256, 0, stream>>>(lamf, lp[p]);

  // Phase 2: 10 masked PGD iterations
  for (int it = 0; it < 10; ++it) {
    gemm_bf16_nt<EPI_PGD><<<dim3(MM / 128, BB / 128), 256, 0, stream>>>(
        lp[p], Qbf, MM, MM, MM, MM, nullptr, lp[1 - p], lamf, cbuf, actb,
        nullptr, nullptr, nullptr);
    p ^= 1;
  }
  // p == 0 again

  // out = u - lam A
  gemm_bf16_nt<EPI_FINAL><<<dim3(NN / 128, BB / 128), 256, 0, stream>>>(
      lp[p], ATbf, MM, MM, MM, NN, out, nullptr, nullptr, nullptr, nullptr,
      nullptr, u, nullptr);
}
